// CapsuleNet_584115552570
// MI455X (gfx1250) — matrix operand order repaired and checked by reading, not compiled, not run on hardware
//
#include <hip/hip_runtime.h>
#include <hip/hip_bf16.h>

typedef __attribute__((ext_vector_type(16))) _Float16 v16h;
typedef __attribute__((ext_vector_type(8)))  _Float16 v8h;
typedef __attribute__((ext_vector_type(8)))  float    v8f;

union Frag { v16h v; v8h h[2]; };

// ---------------------------------------------------------------------------
// One-shot f32 -> f16 conversion (n must be a multiple of 4).
// ---------------------------------------------------------------------------
__global__ __launch_bounds__(256) void cvt_f32_to_f16(
    const float* __restrict__ src, _Float16* __restrict__ dst, int n)
{
  const int i4 = (blockIdx.x * blockDim.x + threadIdx.x) * 4;
  if (i4 < n) {
    const float4 f = *(const float4*)(src + i4);
    dst[i4 + 0] = (_Float16)f.x;
    dst[i4 + 1] = (_Float16)f.y;
    dst[i4 + 2] = (_Float16)f.z;
    dst[i4 + 3] = (_Float16)f.w;
  }
}

// ---------------------------------------------------------------------------
// Implicit-GEMM conv, 9x9 kernel, NCHW, all-f16 data / f32 accumulate.
//   M = O, N = B*OH*OW, K = CIN*81, k = c*81 + r (r = ky*9+kx)
//   -> weight reads CONTIGUOUS in k.
// Block: 256 thr (8 waves), tile 128(M) x 64(N); wave tile 32x32 (4 WMMAs).
// Double-buffered LDS pipeline: the NEXT K-step's A tile is fetched with
// gfx1250 GLOBAL_LOAD_ASYNC_TO_LDS_B128 (ASYNCcnt) and B tile gathered while
// the CURRENT tiles feed the WMMAs; one barrier + s_wait_asynccnt per step.
// ---------------------------------------------------------------------------
template<int STRIDE, bool RELU, int CIN>
__global__ __launch_bounds__(256) void conv_wmma(
    const _Float16* __restrict__ in, const _Float16* __restrict__ wgt,
    const float* __restrict__ bias, _Float16* __restrict__ out,
    int Bn, int H, int Wd, int O, int OH, int OW)
{
  constexpr int Ktot = CIN * 81;
  constexpr bool KVEC = (Ktot % 32) == 0;   // async A staging, no k guards
  const int Ntot = Bn * OH * OW;
  const int bm = blockIdx.x * 128;
  const int bn = blockIdx.y * 64;

  __shared__ _Float16 As[2][128][40];  // [buf][m][k], row stride 80B
  __shared__ _Float16 Bs[2][64][40];   // [buf][n][k]

  const int tid  = threadIdx.x;
  const int lane = tid & 31;
  const int wv   = tid >> 5;
  const int wm   = wv & 3;             // wave M tile: 32*wm
  const int wn   = wv >> 2;            // wave N tile: 32*wn
  const int lo   = lane & 15;
  const int hi   = lane >> 4;

  v8f accs[2][2] = { { {}, {} }, { {}, {} } };

  // A staging: 128 rows x 32 k, 2 threads/row, 16 consecutive k each
  const int arow = tid & 127;
  const int akb  = (tid >> 7) * 16;
  const _Float16* wrow = wgt + (size_t)(bm + arow) * Ktot;

  // B staging: 64 rows x 32 k, 4 threads/row, 8 consecutive k each
  const int brow = tid & 63;
  const int bkb  = (tid >> 6) * 8;
  const int nIdx = bn + brow;
  int ibase = 0, iy0 = 0, ix0 = 0;
  const bool nok = nIdx < Ntot;
  if (nok) {
    const int nb = nIdx / (OH * OW);
    int p2 = nIdx - nb * OH * OW;
    const int noy = p2 / OW;
    const int nox = p2 - noy * OW;
    iy0 = noy * STRIDE;
    ix0 = nox * STRIDE;
    ibase = nb * CIN * H * Wd;
  }

  const int kSteps = (Ktot + 31) >> 5;

  auto stageA = [&](int ks2, int buf) {
    const int kb0 = ks2 << 5;
    if constexpr (KVEC) {
      const _Float16* gsrc = wrow + kb0 + akb;              // 16B-aligned
      const unsigned ldsA = (unsigned)(size_t)&As[buf][arow][akb];
      asm volatile(
          "global_load_async_to_lds_b128 %0, %1, off\n\t"
          "global_load_async_to_lds_b128 %0, %1, off offset:16"
          :: "v"(ldsA), "v"(gsrc) : "memory");
    } else {
      #pragma unroll
      for (int j = 0; j < 16; ++j) {
        const int k = kb0 + akb + j;
        As[buf][arow][akb + j] = (k < Ktot) ? wrow[k] : (_Float16)0.f;
      }
    }
  };

  auto stageB = [&](int ks2, int buf) {
    int k = (ks2 << 5) + bkb;
    int c = k / 81;
    int r = k - c * 81;
    #pragma unroll
    for (int j = 0; j < 8; ++j) {
      _Float16 bv = (_Float16)0.f;
      if (nok && (KVEC || k < Ktot)) {
        const int ky = r / 9, kx = r - ky * 9;
        bv = in[(size_t)ibase + (c * H + iy0 + ky) * Wd + ix0 + kx];
      }
      Bs[buf][brow][bkb + j] = bv;
      ++k; ++r;
      if (r == 81) { r = 0; ++c; }
    }
  };

  // prologue: fill buffer 0
  stageA(0, 0);
  stageB(0, 0);
  if constexpr (KVEC)
    asm volatile("s_wait_asynccnt 0" ::: "memory");
  __syncthreads();

  for (int ksi = 0; ksi < kSteps; ++ksi) {
    const int cur = ksi & 1;
    const int nxt = cur ^ 1;

    // issue next tile's staging (overlaps with WMMAs below)
    if (ksi + 1 < kSteps) {
      stageA(ksi + 1, nxt);
      stageB(ksi + 1, nxt);
    }

    // A frag: lane<16 holds row, K 0..15 contiguous; lane>=16: K 16..31
    Frag a0, a1, b0, b1;
    a0.h[0] = *(const v8h*)&As[cur][wm * 32 + lo][hi * 16];
    a0.h[1] = *(const v8h*)&As[cur][wm * 32 + lo][hi * 16 + 8];
    a1.h[0] = *(const v8h*)&As[cur][wm * 32 + 16 + lo][hi * 16];
    a1.h[1] = *(const v8h*)&As[cur][wm * 32 + 16 + lo][hi * 16 + 8];
    // B frag: lane<16 col n, K 0..15 contiguous; lane>=16 col n, K 16..31
    b0.h[0] = *(const v8h*)&Bs[cur][wn * 32 + lo][hi * 16];
    b0.h[1] = *(const v8h*)&Bs[cur][wn * 32 + lo][hi * 16 + 8];
    b1.h[0] = *(const v8h*)&Bs[cur][wn * 32 + 16 + lo][hi * 16];
    b1.h[1] = *(const v8h*)&Bs[cur][wn * 32 + 16 + lo][hi * 16 + 8];

    accs[0][0] = __builtin_amdgcn_wmma_f32_16x16x32_f16(
        false, a0.v, false, b0.v, (short)0, accs[0][0], false, false);
    accs[0][1] = __builtin_amdgcn_wmma_f32_16x16x32_f16(
        false, a0.v, false, b1.v, (short)0, accs[0][1], false, false);
    accs[1][0] = __builtin_amdgcn_wmma_f32_16x16x32_f16(
        false, a1.v, false, b0.v, (short)0, accs[1][0], false, false);
    accs[1][1] = __builtin_amdgcn_wmma_f32_16x16x32_f16(
        false, a1.v, false, b1.v, (short)0, accs[1][1], false, false);

    // next-tile async A copy must have landed before anyone reads it
    if constexpr (KVEC)
      asm volatile("s_wait_asynccnt 0" ::: "memory");
    __syncthreads();
  }

  // epilogue: C/D layout -> element (m = j + 8*hi, n = lo) per VGPR j
  #pragma unroll
  for (int nj = 0; nj < 2; ++nj) {
    const int n = bn + wn * 32 + nj * 16 + lo;
    if (n < Ntot) {
      const int b = n / (OH * OW);
      int p2 = n - b * OH * OW;
      const int oy = p2 / OW, ox = p2 - oy * OW;
      #pragma unroll
      for (int mi = 0; mi < 2; ++mi) {
        #pragma unroll
        for (int j = 0; j < 8; ++j) {
          const int m = bm + wm * 32 + mi * 16 + hi * 8 + j;
          float vl = accs[mi][nj][j] + bias[m];
          if (RELU) vl = fmaxf(vl, 0.f);
          out[(((size_t)b * O + m) * OH + oy) * OW + ox] = (_Float16)vl;
        }
      }
    }
  }
}

// ---------------------------------------------------------------------------
// Decoder GEMM:  out[n,m] = act( sum_k B[n,k] * A[k,m] + bias[m] )
// A = weight [K,M] (f32), B = activations [N,K] (f32).  Tile 64(M) x 32(N),
// 8 waves (4 M x 2 N), one 16x16x32 WMMA per wave per K step.
// ACT: 0=none 1=relu 2=sigmoid
// ---------------------------------------------------------------------------
template<int ACT>
__global__ __launch_bounds__(256) void gemm_wmma(
    const float* __restrict__ A, const float* __restrict__ Bm,
    const float* __restrict__ bias, float* __restrict__ out,
    int M, int N, int K)
{
  const int bm = blockIdx.x * 64;
  const int bn = blockIdx.y * 32;

  __shared__ _Float16 As[64][40];
  __shared__ _Float16 Bs[32][40];

  const int tid  = threadIdx.x;
  const int lane = tid & 31;
  const int wv   = tid >> 5;
  const int wm   = wv & 3;
  const int wn   = wv >> 2;
  const int lo   = lane & 15;
  const int hi   = lane >> 4;

  v8f acc = {};

  const int am   = tid & 63;
  const int akb  = (tid >> 6) * 8;
  const int bn_r = tid >> 3;        // 32 rows, 8 threads each
  const int bkb  = (tid & 7) * 4;

  for (int k0 = 0; k0 < K; k0 += 32) {
    #pragma unroll
    for (int j = 0; j < 8; ++j) {
      const int k = k0 + akb + j;
      As[am][akb + j] = (_Float16)((k < K) ? A[(size_t)k * M + bm + am] : 0.f);
    }
    #pragma unroll
    for (int j = 0; j < 4; ++j) {
      const int kk = bkb + j;
      const int k  = k0 + kk;
      const int n  = bn + bn_r;
      float bv = 0.f;
      if (k < K && n < N) bv = Bm[(size_t)n * K + k];
      Bs[bn_r][kk] = (_Float16)bv;
    }
    __syncthreads();

    Frag a, b;
    a.h[0] = *(const v8h*)&As[wm * 16 + lo][hi * 16];
    a.h[1] = *(const v8h*)&As[wm * 16 + lo][hi * 16 + 8];
    b.h[0] = *(const v8h*)&Bs[wn * 16 + lo][hi * 16];
    b.h[1] = *(const v8h*)&Bs[wn * 16 + lo][hi * 16 + 8];
    acc = __builtin_amdgcn_wmma_f32_16x16x32_f16(
        false, a.v, false, b.v, (short)0, acc, false, false);
    __syncthreads();
  }

  const int n = bn + wn * 16 + lo;
  if (n < N) {
    #pragma unroll
    for (int j = 0; j < 8; ++j) {
      const int m = bm + wm * 16 + hi * 8 + j;
      float vl = acc[j] + bias[m];
      if (ACT == 1) vl = fmaxf(vl, 0.f);
      if (ACT == 2) vl = 1.f / (1.f + __expf(-vl));
      out[(size_t)n * M + m] = vl;
    }
  }
}

// ---------------------------------------------------------------------------
// Primary-capsule reshape + squash:  f16 [B,256,6,6] -> f32 p[B,1152,8]
//   i = ch/32, r = (ch%32)*36 + y*6 + x, squash over the 8-vector.
// ---------------------------------------------------------------------------
__global__ __launch_bounds__(256) void prim_squash(
    const _Float16* __restrict__ a2, float* __restrict__ p)
{
  const int idx = blockIdx.x * blockDim.x + threadIdx.x;  // b*1152 + r
  if (idx >= 32 * 1152) return;
  const int b = idx / 1152, r = idx - b * 1152;
  const int ch32 = r / 36, pix = r - ch32 * 36;
  float t[8];
  float sn = 0.f;
  #pragma unroll
  for (int i = 0; i < 8; ++i) {
    const int ch = i * 32 + ch32;
    const float vv = (float)a2[((size_t)b * 256 + ch) * 36 + pix];
    t[i] = vv;
    sn += vv * vv;
  }
  const float scale = sqrtf(sn) / (1.f + sn);
  #pragma unroll
  for (int i = 0; i < 8; ++i) p[(size_t)idx * 8 + i] = t[i] * scale;
}

// ---------------------------------------------------------------------------
// Dynamic routing.  One workgroup per (class c, batch b).
//   x: [B,R,Ci], W: [C,R,Ci,16], out: [B,C,16].  Priors cached in LDS (f16).
// ---------------------------------------------------------------------------
__global__ __launch_bounds__(256) void routing_kernel(
    const float* __restrict__ x, const float* __restrict__ Wt,
    float* __restrict__ out, int R, int Ci)
{
  extern __shared__ unsigned char smraw[];
  float*    logits = (float*)smraw;          // R
  float*    probs  = logits + R;             // R
  float*    red    = probs + R;              // 256
  float*    sv     = red + 256;              // 16
  float*    vout   = sv + 16;                // 16
  _Float16* priors = (_Float16*)(vout + 16); // R*16

  const int c   = blockIdx.x;
  const int b   = blockIdx.y;
  const int C   = gridDim.x;
  const int tid = threadIdx.x;
  const int o   = tid & 15;
  const int rg  = tid >> 4;

  const float* xb = x  + (size_t)b * R * Ci;
  const float* Wc = Wt + (size_t)c * R * Ci * 16;

  // priors[r,o] = sum_i x[b,r,i] * W[c,r,i,o]
  for (int idx = tid; idx < R * 16; idx += 256) {
    const int r = idx >> 4, oo = idx & 15;
    float s = 0.f;
    for (int i = 0; i < Ci; ++i)
      s += xb[r * Ci + i] * Wc[(size_t)(r * Ci + i) * 16 + oo];
    priors[idx] = (_Float16)s;
  }
  for (int r = tid; r < R; r += 256) logits[r] = 0.f;
  __syncthreads();

  for (int it = 0; it < 3; ++it) {
    // softmax over route nodes r: max
    float mx = -3.0e38f;
    for (int r = tid; r < R; r += 256) mx = fmaxf(mx, logits[r]);
    red[tid] = mx; __syncthreads();
    for (int s2 = 128; s2 > 0; s2 >>= 1) {
      if (tid < s2) red[tid] = fmaxf(red[tid], red[tid + s2]);
      __syncthreads();
    }
    mx = red[0]; __syncthreads();

    // exp + partition Z
    float zp = 0.f;
    for (int r = tid; r < R; r += 256) {
      const float e = __expf(logits[r] - mx);
      probs[r] = e;
      zp += e;
    }
    red[tid] = zp; __syncthreads();
    for (int s2 = 128; s2 > 0; s2 >>= 1) {
      if (tid < s2) red[tid] += red[tid + s2];
      __syncthreads();
    }
    const float Z = red[0]; __syncthreads();

    // s[o] = (1/Z) sum_r probs[r] * priors[r,o]   (16 o-groups of 16 threads)
    float part = 0.f;
    for (int r = rg; r < R; r += 16)
      part += probs[r] * (float)priors[r * 16 + o];
    red[tid] = part; __syncthreads();
    for (int s2 = 128; s2 >= 16; s2 >>= 1) {
      if (tid < s2) red[tid] += red[tid + s2];
      __syncthreads();
    }
    if (tid < 16) sv[tid] = red[tid] / Z;
    __syncthreads();

    // squash
    if (tid == 0) {
      float sn = 0.f;
      for (int oo = 0; oo < 16; ++oo) sn += sv[oo] * sv[oo];
      const float scale = sqrtf(sn) / (1.f + sn);
      for (int oo = 0; oo < 16; ++oo) vout[oo] = sv[oo] * scale;
    }
    __syncthreads();

    if (it < 2) {
      for (int r = tid; r < R; r += 256) {
        float acc2 = 0.f;
        #pragma unroll
        for (int oo = 0; oo < 16; ++oo)
          acc2 += (float)priors[r * 16 + oo] * vout[oo];
        logits[r] += acc2;
      }
      __syncthreads();
    }
  }
  if (tid < 16) out[((size_t)b * C + c) * 16 + tid] = vout[tid];
}

// ---------------------------------------------------------------------------
// Class head: norms -> softmax (to d_out) -> argmax -> masked z [B,160]
// ---------------------------------------------------------------------------
__global__ __launch_bounds__(256) void class_head(
    const float* __restrict__ v, float* __restrict__ classes,
    float* __restrict__ z)
{
  const int b = blockIdx.x;
  __shared__ float norms[16];
  __shared__ int best;
  const int tid = threadIdx.x;
  if (tid < 10) {
    float sn = 0.f;
    for (int oo = 0; oo < 16; ++oo) {
      const float t = v[((size_t)b * 10 + tid) * 16 + oo];
      sn += t * t;
    }
    norms[tid] = sqrtf(sn);
  }
  __syncthreads();
  if (tid == 0) {
    float mx = norms[0]; int bi = 0;
    for (int c2 = 1; c2 < 10; ++c2)
      if (norms[c2] > mx) { mx = norms[c2]; bi = c2; }
    float e[10], Z = 0.f;
    for (int c2 = 0; c2 < 10; ++c2) { e[c2] = __expf(norms[c2] - mx); Z += e[c2]; }
    for (int c2 = 0; c2 < 10; ++c2) classes[(size_t)b * 10 + c2] = e[c2] / Z;
    best = bi;
  }
  __syncthreads();
  if (tid < 160) {
    const int c2 = tid >> 4;
    z[(size_t)b * 160 + tid] =
        (c2 == best) ? v[((size_t)b * 10 + c2) * 16 + (tid & 15)] : 0.f;
  }
}

// ---------------------------------------------------------------------------
extern "C" void kernel_launch(void* const* d_in, const int* in_sizes, int n_in,
                              void* d_out, int out_size, void* d_ws, size_t ws_size,
                              hipStream_t stream) {
  (void)in_sizes; (void)n_in; (void)out_size; (void)ws_size;

  const float* x       = (const float*)d_in[0];
  const float* conv0_w = (const float*)d_in[1];
  const float* conv0_b = (const float*)d_in[2];
  const float* conv1_w = (const float*)d_in[3];
  const float* conv1_b = (const float*)d_in[4];
  const float* prim_w  = (const float*)d_in[5];
  const float* prim_b  = (const float*)d_in[6];
  const float* W1      = (const float*)d_in[7];
  const float* W2      = (const float*)d_in[8];
  const float* dec_w1  = (const float*)d_in[9];
  const float* dec_b1  = (const float*)d_in[10];
  const float* dec_w2  = (const float*)d_in[11];
  const float* dec_b2  = (const float*)d_in[12];
  const float* dec_w3  = (const float*)d_in[13];
  const float* dec_b3  = (const float*)d_in[14];

  // ---- workspace: f16 region first, then f32 region ----
  _Float16* xh  = (_Float16*)d_ws;   // 119072   = 32*1*61*61
  _Float16* w0h = xh  + 119072;      // 20736    = 256*81
  _Float16* w1h = w0h + 20736;       // 5308416  = 256*256*81
  _Float16* wph = w1h + 5308416;     // 5308416
  _Float16* a0h = wph + 5308416;     // 5971968  = 32*256*27*27
  _Float16* a1h = a0h + 5971968;     // 2957312  = 32*256*19*19
  _Float16* a2h = a1h + 2957312;     // 294912   = 32*256*6*6
  float* fbase  = (float*)(a2h + 294912);
  float* p   = fbase;           // [32,1152,8]  = 294912
  float* u   = p   + 294912;    // [32,64,16]   = 32768
  float* vv  = u   + 32768;     // [32,10,16]   = 5120
  float* z   = vv  + 5120;      // [32,160]     = 5120
  float* dl1 = z   + 5120;      // [32,1024]    = 32768
  float* dl2 = dl1 + 32768;     // [32,2048]    = 65536

  float* classes = (float*)d_out;        // [32,10]
  float* recon   = classes + 320;        // [32,4096]

  // ---- one-shot f32 -> f16 conversions (inputs + weights) ----
  cvt_f32_to_f16<<<(119072 / 4 + 255) / 256, 256, 0, stream>>>(x, xh, 119072);
  cvt_f32_to_f16<<<(20736 / 4 + 255) / 256, 256, 0, stream>>>(conv0_w, w0h, 20736);
  cvt_f32_to_f16<<<(5308416 / 4 + 255) / 256, 256, 0, stream>>>(conv1_w, w1h, 5308416);
  cvt_f32_to_f16<<<(5308416 / 4 + 255) / 256, 256, 0, stream>>>(prim_w, wph, 5308416);

  // conv0: [32,1,61,61] -> [32,256,27,27], relu   (N=23328, K=81)
  conv_wmma<2, true, 1><<<dim3(2, (23328 + 63) / 64), 256, 0, stream>>>(
      xh, w0h, conv0_b, a0h, 32, 61, 61, 256, 27, 27);
  // conv1: -> [32,256,19,19], relu                (N=11552, K=20736)
  conv_wmma<1, true, 256><<<dim3(2, (11552 + 63) / 64), 256, 0, stream>>>(
      a0h, w1h, conv1_b, a1h, 32, 27, 27, 256, 19, 19);
  // prim:  -> [32,256,6,6], no relu               (N=1152)
  conv_wmma<2, false, 256><<<dim3(2, (1152 + 63) / 64), 256, 0, stream>>>(
      a1h, wph, prim_b, a2h, 32, 19, 19, 256, 6, 6);

  prim_squash<<<(32 * 1152 + 255) / 256, 256, 0, stream>>>(a2h, p);

  {
    const int R = 1152, Ci = 8;
    const size_t shm = (size_t)(R + R + 256 + 32) * 4 + (size_t)R * 16 * 2;
    routing_kernel<<<dim3(64, 32), 256, shm, stream>>>(p, W1, u, R, Ci);
  }
  {
    const int R = 64, Ci = 16;
    const size_t shm = (size_t)(R + R + 256 + 32) * 4 + (size_t)R * 16 * 2;
    routing_kernel<<<dim3(10, 32), 256, shm, stream>>>(u, W2, vv, R, Ci);
  }

  class_head<<<32, 256, 0, stream>>>(vv, classes, z);

  gemm_wmma<1><<<dim3(1024 / 64, 1), 256, 0, stream>>>(dec_w1, z,   dec_b1, dl1, 1024, 32, 160);
  gemm_wmma<1><<<dim3(2048 / 64, 1), 256, 0, stream>>>(dec_w2, dl1, dec_b2, dl2, 2048, 32, 1024);
  gemm_wmma<2><<<dim3(4096 / 64, 1), 256, 0, stream>>>(dec_w3, dl2, dec_b3, recon, 4096, 32, 2048);
}
